// Conv2d_85426899517785
// MI455X (gfx1250) — matrix operand order repaired and checked by reading, not compiled, not run on hardware
//
#include <hip/hip_runtime.h>

typedef __attribute__((ext_vector_type(16))) __bf16       v16bf;
typedef __attribute__((ext_vector_type(8)))  float        v8f;
typedef __attribute__((ext_vector_type(4)))  float        f32x4;
typedef __attribute__((ext_vector_type(4)))  unsigned int u32x4;
typedef unsigned short ush;

#define CIN      128
#define COUT     256
#define HH       56
#define WWID     56
#define HW       3136            // 56*56
#define KTOT     1152            // 128*9
#define STOT     100352          // 32*3136 = 784*128, exact tiling
#define CHW_IN   (CIN * HW)      // 401408
#define CHW_OUT  (COUT * HW)     // 802816
#define XELEMS   (32 * CHW_IN)   // 12,845,056
#define WELEMS   (COUT * KTOT)   // 294,912
#define XBYTES   (XELEMS * 2)    // bf16 staged x

#define MT  128
#define NT  128
#define LDA 40   // LDS row stride in halves: 32 + 8 pad (80 B, keeps 16B alignment)

__device__ __forceinline__ ush f2bf(float f) {
  unsigned int u = __builtin_bit_cast(unsigned int, f);
  u += 0x7FFFu + ((u >> 16) & 1u);          // round-to-nearest-even
  return (ush)(u >> 16);
}

union Frag {
  v16bf v;
  u32x4 q[2];
};

// ---- pre-pass 1: x f32 -> bf16, straight layout. 8 elems/thread, exact grid ----
__global__ __launch_bounds__(256)
void cvt_x_bf16(const float* __restrict__ x, ush* __restrict__ xbf) {
  const long long i = ((long long)blockIdx.x * 256 + threadIdx.x) * 8;
  f32x4 a = *(const f32x4*)(x + i);
  f32x4 b = *(const f32x4*)(x + i + 4);
  u32x4 pk = {(unsigned)f2bf(a[0]) | ((unsigned)f2bf(a[1]) << 16),
              (unsigned)f2bf(a[2]) | ((unsigned)f2bf(a[3]) << 16),
              (unsigned)f2bf(b[0]) | ((unsigned)f2bf(b[1]) << 16),
              (unsigned)f2bf(b[2]) | ((unsigned)f2bf(b[3]) << 16)};
  *(u32x4*)(xbf + i) = pk;
}

// ---- pre-pass 2: w f32 [co][ci][kh][kw] -> bf16 [co][ (kh*3+kw)*128 + ci ] ----
__global__ __launch_bounds__(256)
void cvt_w_bf16(const float* __restrict__ w, ush* __restrict__ wbf) {
  const int o   = blockIdx.x * 256 + threadIdx.x;   // 0..294911, exact grid
  const int co  = o / KTOT;
  const int rem = o - co * KTOT;
  const int r   = rem >> 7;                          // kh*3+kw
  const int ci  = rem & 127;
  wbf[o] = f2bf(w[co * KTOT + ci * 9 + r]);
}

// ---- main GEMM: im2col(bf16 x) @ wbf via v_wmma_f32_16x16x32_bf16 ----
__global__ __launch_bounds__(256)
void conv2d_wmma_bf16_main(const ush* __restrict__ xbf,
                           const ush* __restrict__ wbf,
                           float* __restrict__ out) {
  __shared__ ush Bs[2][NT][LDA];   // double-buffered im2col tile [s_local][k]

  const int tid  = threadIdx.x;
  const int lane = tid & 31;
  const int wv   = tid >> 5;            // 0..7 waves
  const int wm   = wv >> 1;             // 0..3 : 32 M-rows per wave
  const int wn   = wv & 1;              // 0..1 : 64 N-cols per wave

  const int mBase = blockIdx.y * MT;    // 0 or 128
  const int sBase = blockIdx.x * NT;

  // ---- B staging mapping: one spatial position per thread, 16 ci's per half ----
  const int sLoc  = tid >> 1;           // 0..127
  const int half  = tid & 1;
  const int s     = sBase + sLoc;
  const int n_img = s / HW;
  const int hw    = s - n_img * HW;
  const int oh    = hw / WWID;
  const int ow    = hw - oh * WWID;
  const ush* xThread = xbf + (long long)n_img * CHW_IN + (long long)(half * 16) * HW;

  // ---- WMMA fragment lane geometry (ISA 7.12.2, 16-bit operands, wave32) ----
  const int lhi = lane >> 4;
  const int l15 = lane & 15;
  const int aKb = lhi ? 8 : 0;          // A: halves [aKb..+7] and [16+aKb..+7]
  const int bKs = lhi ? 8 : 0;          // B: halves [bKs..+7] and [16+bKs..+7]

  // A fragment base pointers (direct global, weights are L2-resident)
  const ush* aRow0 = wbf + (mBase + wm * 32 + 0 * 16 + l15) * KTOT;
  const ush* aRow1 = wbf + (mBase + wm * 32 + 1 * 16 + l15) * KTOT;

  v8f acc[2][4];
  {
    v8f z = {};
#pragma unroll
    for (int i = 0; i < 2; ++i)
#pragma unroll
      for (int j = 0; j < 4; ++j) acc[i][j] = z;
  }

  int it = 0;
  for (int r = 0; r < 9; ++r) {         // (kh,kw) uniform per chunk group
    const int kh = r / 3;
    const int kw = r - kh * 3;
    const int ih = oh + kh - 1;
    const int iw = ow + kw - 1;
    const bool valid =
        ((unsigned)ih < (unsigned)HH) & ((unsigned)iw < (unsigned)WWID);
    // Branchless halo: clamp offset in-bounds (one cndmask), zero via AND mask.
    const int          off   = valid ? (ih * WWID + iw) : 0;
    const unsigned int vmask = valid ? 0xFFFFFFFFu : 0u;
    const ush*         xr    = xThread + off;

    for (int cb = 0; cb < 4; ++cb) {    // ci blocks of 32
      // ---- stage B tile chunk: 16 unconditional constant-stride gathers ----
      {
        const ush* xp = xr + cb * 32 * HW;
        unsigned int packed[8];
#pragma unroll
        for (int p = 0; p < 8; ++p) {
          unsigned int lo = (unsigned int)xp[(2 * p) * HW];
          unsigned int hi = (unsigned int)xp[(2 * p + 1) * HW];
          packed[p] = (lo | (hi << 16)) & vmask;
        }
        u32x4* dst = (u32x4*)&Bs[it & 1][sLoc][half * 16];
        u32x4 q0 = {packed[0], packed[1], packed[2], packed[3]};
        u32x4 q1 = {packed[4], packed[5], packed[6], packed[7]};
        dst[0] = q0;
        dst[1] = q1;
      }
      __syncthreads();   // single barrier/iter: double buffer protects next stage

      // ---- fragments: A direct from global (contiguous 16B runs), B from LDS ----
      const int kBase = r * 128 + cb * 32;
      Frag a[2], b[4];
      a[0].q[0] = *(const u32x4*)(aRow0 + kBase + aKb);
      a[0].q[1] = *(const u32x4*)(aRow0 + kBase + 16 + aKb);
      a[1].q[0] = *(const u32x4*)(aRow1 + kBase + aKb);
      a[1].q[1] = *(const u32x4*)(aRow1 + kBase + 16 + aKb);
#pragma unroll
      for (int ni = 0; ni < 4; ++ni) {
        const int rr = wn * 64 + ni * 16 + l15;
        b[ni].q[0] = *(const u32x4*)&Bs[it & 1][rr][bKs];
        b[ni].q[1] = *(const u32x4*)&Bs[it & 1][rr][bKs + 16];
      }
#pragma unroll
      for (int mi = 0; mi < 2; ++mi)
#pragma unroll
        for (int ni = 0; ni < 4; ++ni)
          acc[mi][ni] = __builtin_amdgcn_wmma_f32_16x16x32_bf16(
              false, a[mi].v, false, b[ni].v, (short)0, acc[mi][ni], false, false);
      ++it;
    }
  }

  // ---- epilogue: C/D layout (VGPR v -> M, lane&15 -> N) -> NCHW, NT stores ----
#pragma unroll
  for (int ni = 0; ni < 4; ++ni) {
    const int sOut = sBase + wn * 64 + ni * 16 + l15;
    const int nI   = sOut / HW;
    const int hwI  = sOut - nI * HW;
    const long long base = (long long)nI * CHW_OUT + hwI;
#pragma unroll
    for (int mi = 0; mi < 2; ++mi) {
#pragma unroll
      for (int v = 0; v < 8; ++v) {
        const int co = mBase + wm * 32 + mi * 16 + (lhi ? 8 : 0) + v;
        __builtin_nontemporal_store(acc[mi][ni][v], out + base + (long long)co * HW);
      }
    }
  }
}

extern "C" void kernel_launch(void* const* d_in, const int* in_sizes, int n_in,
                              void* d_out, int out_size, void* d_ws, size_t ws_size,
                              hipStream_t stream) {
  const float* x   = (const float*)d_in[0];   // [32,128,56,56] f32
  const float* w   = (const float*)d_in[1];   // [256,128,3,3]  f32
  float*       out = (float*)d_out;           // [32,256,56,56] f32

  ush* xbf = (ush*)d_ws;                        // 25,690,112 B
  ush* wbf = (ush*)((char*)d_ws + XBYTES);      // +589,824 B  (ws >= ~26.3 MB)

  // pre-passes (exact grids, no tails)
  hipLaunchKernelGGL(cvt_x_bf16, dim3(XELEMS / (256 * 8)), dim3(256), 0, stream, x, xbf);
  hipLaunchKernelGGL(cvt_w_bf16, dim3(WELEMS / 256), dim3(256), 0, stream, w, wbf);

  // main GEMM: (784 N-tiles) x (2 M-tiles), 8 wave32 waves per block
  hipLaunchKernelGGL(conv2d_wmma_bf16_main, dim3(STOT / NT, COUT / MT, 1),
                     dim3(256), 0, stream, xbf, wbf, out);
}
